// AttentionGCN_44633300140824
// MI455X (gfx1250) — matrix operand order repaired and checked by reading, not compiled, not run on hardware
//
#include <hip/hip_runtime.h>
#include <hip/hip_bf16.h>
#include <stdint.h>

typedef __attribute__((ext_vector_type(16))) __bf16 v16bf;
typedef __attribute__((ext_vector_type(8)))  float  v8f;

#define NGRAPHS 8192
#define CDIV(a,b) (((a)+(b)-1)/(b))

// ---------------------------------------------------------------- utilities
__global__ void fill_f32(float* p, float v, long n) {
  long i = (long)blockIdx.x * blockDim.x + threadIdx.x;
  long s = (long)gridDim.x * blockDim.x;
  for (; i < n; i += s) p[i] = v;
}

__global__ void edge_deg(const int* __restrict__ dst, float* __restrict__ deg, int E) {
  int e = blockIdx.x * blockDim.x + threadIdx.x;
  if (e < E) atomicAdd(&deg[dst[e]], 1.0f);
}

__global__ void deg_to_dinv(float* deg, int n) {
  int i = blockIdx.x * blockDim.x + threadIdx.x;
  if (i < n) { float d = deg[i]; deg[i] = d > 0.f ? rsqrtf(d) : 0.f; }
}

// norm[e] = dinv[src]*dinv[dst] precomputed once (6 random gathers -> 1 stream)
__global__ void edge_norm(const int* __restrict__ src, const int* __restrict__ dst,
                          const float* __restrict__ dinv, float* __restrict__ nrm, int E) {
  int e = blockIdx.x * blockDim.x + threadIdx.x;
  if (e < E) nrm[e] = dinv[src[e]] * dinv[dst[e]];
}

// layer-1 "GEMM": x[N,1] @ W1[1,64]  -> t[N,64]
__global__ void l1_transform(const float* __restrict__ x, const float* __restrict__ W1,
                             float* __restrict__ t, int n) {
  int i = blockIdx.x * blockDim.x + threadIdx.x;
  if (i < n * 64) { int node = i >> 6, j = i & 63; t[i] = x[node] * W1[j]; }
}

// agg = t * dinv[node]^2   (self-loop contribution; also initializes agg)
__global__ void self_init(const float* __restrict__ t, const float* __restrict__ dinv,
                          float* __restrict__ agg, int total, int sh) {
  int i = blockIdx.x * blockDim.x + threadIdx.x;
  if (i < total) { float di = dinv[i >> sh]; agg[i] = t[i] * di * di; }
}

// agg[dst] += t[src] * norm[e], one float4 of channels per thread
__global__ void edge_scatter(const float* __restrict__ t, float* __restrict__ agg,
                             const int* __restrict__ src, const int* __restrict__ dst,
                             const float* __restrict__ enorm,
                             long total, int sh, int c4mask, int C) {
  long i = (long)blockIdx.x * blockDim.x + threadIdx.x;
  if (i >= total) return;
  int e  = (int)(i >> sh);
  int c4 = ((int)i) & c4mask;
  int s = src[e], d = dst[e];
  float nrm = enorm[e];
  float4 v = *(const float4*)(t + (size_t)s * C + c4 * 4);
  float* ap = agg + (size_t)d * C + c4 * 4;
  atomicAdd(ap + 0, v.x * nrm);
  atomicAdd(ap + 1, v.y * nrm);
  atomicAdd(ap + 2, v.z * nrm);
  atomicAdd(ap + 3, v.w * nrm);
}

// ------------------------------------------------- WMMA bf16 GEMM (f32 acc)
// C[M,Nc] = act(A)[M,K] @ W[K,Nc];  act = relu(A + a_bias[k]) if a_bias else A.
// Block tile 64x64, BK=32, 256 threads = 8 waves (4x2 grid); each wave owns a
// 16x32 strip of C (2 accumulators, A fragment reused across 2 WMMAs).
// Requires M%64==0, Nc%64==0, K%32==0 (all shapes here satisfy this).
__global__ __launch_bounds__(256) void wmma_gemm_bf16(
    const float* __restrict__ A, const float* __restrict__ W,
    float* __restrict__ C, int M, int Nc, int K,
    const float* __restrict__ a_bias) {
  __shared__ __bf16 As[64][34];   // even pad: 4B-aligned pairs, fewer bank conflicts
  __shared__ __bf16 Bs[32][66];

  const int tid  = threadIdx.x;
  const int lane = tid & 31, wave = tid >> 5;
  const int mt  = wave & 3;             // 4 M-tiles
  const int nb0 = (wave >> 2) * 32;     // 2 N-strips of 32
  const int mBase = blockIdx.x * 64;
  const int nBase = blockIdx.y * 64;

  v8f acc0 = {0.f, 0.f, 0.f, 0.f, 0.f, 0.f, 0.f, 0.f};
  v8f acc1 = {0.f, 0.f, 0.f, 0.f, 0.f, 0.f, 0.f, 0.f};

  const int arow = tid >> 2;        // 0..63
  const int acol = (tid & 3) * 8;   // 0,8,16,24
  const int brow = tid >> 3;        // 0..31
  const int bcol = (tid & 7) * 8;   // 0..56

  for (int k0 = 0; k0 < K; k0 += 32) {
    __syncthreads();
    { // A tile: fp32 global -> (fused bias+relu) -> bf16 LDS
      const float* ap = A + (size_t)(mBase + arow) * K + (k0 + acol);
      float4 v0 = *(const float4*)ap;
      float4 v1 = *(const float4*)(ap + 4);
      float vv[8] = {v0.x, v0.y, v0.z, v0.w, v1.x, v1.y, v1.z, v1.w};
      if (a_bias) {
#pragma unroll
        for (int i = 0; i < 8; ++i) {
          float t = vv[i] + a_bias[k0 + acol + i];
          vv[i] = t > 0.f ? t : 0.f;
        }
      }
#pragma unroll
      for (int i = 0; i < 8; ++i) As[arow][acol + i] = (__bf16)vv[i];
    }
    { // B tile (32 K-rows x 64 N-cols)
      const float* bp = W + (size_t)(k0 + brow) * Nc + (nBase + bcol);
      float4 v0 = *(const float4*)bp;
      float4 v1 = *(const float4*)(bp + 4);
      Bs[brow][bcol + 0] = (__bf16)v0.x;
      Bs[brow][bcol + 1] = (__bf16)v0.y;
      Bs[brow][bcol + 2] = (__bf16)v0.z;
      Bs[brow][bcol + 3] = (__bf16)v0.w;
      Bs[brow][bcol + 4] = (__bf16)v1.x;
      Bs[brow][bcol + 5] = (__bf16)v1.y;
      Bs[brow][bcol + 6] = (__bf16)v1.z;
      Bs[brow][bcol + 7] = (__bf16)v1.w;
    }
    __syncthreads();

    // A fragment (16x32 bf16): lane = m + 16*half; VGPR j holds K pair
    //   k = 16*(j/4) + 8*half + 2*(j%4)
    v16bf afrag, bfrag0, bfrag1;
    {
      const int am = mt * 16 + (lane & 15);
      const int hl = lane >> 4;
#pragma unroll
      for (int j = 0; j < 8; ++j) {
        int k = ((j >> 2) << 4) + hl * 8 + ((j & 3) << 1);
        afrag[2 * j]     = As[am][k];
        afrag[2 * j + 1] = As[am][k + 1];
      }
    }
    // B fragments (32x16 bf16): lane = n + 16*half; VGPR j holds K pair
    //   k = 16*(j/4) + 8*half + 2*(j%4)
    {
      const int bn = lane & 15;
      const int bh = lane >> 4;
#pragma unroll
      for (int j = 0; j < 8; ++j) {
        int k = ((j >> 2) << 4) + bh * 8 + ((j & 3) << 1);
        bfrag0[2 * j]     = Bs[k][nb0 + bn];
        bfrag0[2 * j + 1] = Bs[k + 1][nb0 + bn];
        bfrag1[2 * j]     = Bs[k][nb0 + 16 + bn];
        bfrag1[2 * j + 1] = Bs[k + 1][nb0 + 16 + bn];
      }
    }
    acc0 = __builtin_amdgcn_wmma_f32_16x16x32_bf16(false, afrag, false, bfrag0,
                                                   (short)0, acc0, false, false);
    acc1 = __builtin_amdgcn_wmma_f32_16x16x32_bf16(false, afrag, false, bfrag1,
                                                   (short)0, acc1, false, false);
  }

  // C layout: lane -> n = lane%16; VGPR i -> m = i + 8*(lane/16)
  const int n0 = nBase + nb0 + (lane & 15);
  const int hl = lane >> 4;
#pragma unroll
  for (int i = 0; i < 8; ++i) {
    int m = mBase + mt * 16 + i + hl * 8;
    C[(size_t)m * Nc + n0]      = acc0[i];
    C[(size_t)m * Nc + n0 + 16] = acc1[i];
  }
}

// out[m] = sum_k relu(A[m,k]+bias[k]) * w[k] + b0[0]; one wave per row
__global__ void rowdot_relu(const float* __restrict__ A, const float* __restrict__ bias,
                            const float* __restrict__ w, const float* __restrict__ b0,
                            float* __restrict__ out, int M, int K) {
  int row  = blockIdx.x * 8 + (threadIdx.x >> 5);
  int lane = threadIdx.x & 31;
  if (row >= M) return;
  const float* ar = A + (size_t)row * K;
  float s = 0.f;
  for (int k = lane; k < K; k += 32) {
    float v = ar[k] + bias[k];
    v = v > 0.f ? v : 0.f;
    s += v * w[k];
  }
#pragma unroll
  for (int off = 16; off; off >>= 1) s += __shfl_xor(s, off, 32);
  if (lane == 0) out[row] = s + b0[0];
}

// ------------------------------------------- segment softmax + attention pool
__global__ void seg_max(const float* __restrict__ gate, const int* __restrict__ batch,
                        float* __restrict__ m, int n) {
  int i = blockIdx.x * blockDim.x + threadIdx.x;
  if (i < n) atomicMax(&m[batch[i]], gate[i]);
}

__global__ void exp_z(float* __restrict__ gate, const int* __restrict__ batch,
                      const float* __restrict__ m, float* __restrict__ z, int n) {
  int i = blockIdx.x * blockDim.x + threadIdx.x;
  if (i < n) {
    int b = batch[i];
    float e = expf(gate[i] - m[b]);
    gate[i] = e;
    atomicAdd(&z[b], e);
  }
}

// pooled[b] += (e/z[b]) * relu(agg3[n]+b3)   (h3 computed on the fly)
__global__ void pooled_scatter(const float* __restrict__ agg3, const float* __restrict__ b3,
                               const float* __restrict__ e, const float* __restrict__ z,
                               const int* __restrict__ batch, float* __restrict__ pooled,
                               int n) {
  int i = blockIdx.x * blockDim.x + threadIdx.x;
  if (i >= n * 64) return;
  int node = i >> 6, c4 = i & 63;
  int b = batch[node];
  float wgt = e[node] / z[b];
  float4 v  = *(const float4*)(agg3 + (size_t)node * 256 + c4 * 4);
  float4 bb = *(const float4*)(b3 + c4 * 4);
  float r0 = fmaxf(v.x + bb.x, 0.f) * wgt;
  float r1 = fmaxf(v.y + bb.y, 0.f) * wgt;
  float r2 = fmaxf(v.z + bb.z, 0.f) * wgt;
  float r3 = fmaxf(v.w + bb.w, 0.f) * wgt;
  float* pp = pooled + (size_t)b * 256 + c4 * 4;
  atomicAdd(pp + 0, r0);
  atomicAdd(pp + 1, r1);
  atomicAdd(pp + 2, r2);
  atomicAdd(pp + 3, r3);
}

// ---------------------------------------------------------------------------
extern "C" void kernel_launch(void* const* d_in, const int* in_sizes, int n_in,
                              void* d_out, int out_size, void* d_ws, size_t ws_size,
                              hipStream_t stream) {
  const float* x    = (const float*)d_in[0];
  const int*   eidx = (const int*)d_in[1];
  const int*   batch= (const int*)d_in[2];
  const float* W1 = (const float*)d_in[3];  const float* b1  = (const float*)d_in[4];
  const float* W2 = (const float*)d_in[5];  const float* b2  = (const float*)d_in[6];
  const float* W3 = (const float*)d_in[7];  const float* b3  = (const float*)d_in[8];
  const float* g1 = (const float*)d_in[9];  const float* gb1 = (const float*)d_in[10];
  const float* g2 = (const float*)d_in[11]; const float* gb2 = (const float*)d_in[12];
  const float* g3 = (const float*)d_in[13]; const float* gb3 = (const float*)d_in[14];
  const float* h1 = (const float*)d_in[15]; const float* hb1 = (const float*)d_in[16];
  const float* h2 = (const float*)d_in[17]; const float* hb2 = (const float*)d_in[18];
  const float* h3 = (const float*)d_in[19]; const float* hb3 = (const float*)d_in[20];
  float* out = (float*)d_out;

  const int N = in_sizes[0];          // 200000 (multiple of 64)
  const int E = in_sizes[1] / 2;      // 800000
  const int* esrc = eidx;
  const int* edst = eidx + E;

  // ------- workspace carve (all 256B aligned)
  char* wp = (char*)d_ws;
  auto carve = [&](size_t bytes) -> float* {
    float* p = (float*)wp;
    wp += (bytes + 255) & ~(size_t)255;
    return p;
  };
  float* dinv   = carve((size_t)N * 4);
  float* enorm  = carve((size_t)E * 4);
  float* P      = carve((size_t)N * 256 * 4);
  float* Q      = carve((size_t)N * 256 * 4);
  float* R      = carve((size_t)N * 256 * 4);
  float* ge     = carve((size_t)N * 4);
  float* mbuf   = carve((size_t)NGRAPHS * 4);
  float* zbuf   = carve((size_t)NGRAPHS * 4);
  float* pooled = carve((size_t)NGRAPHS * 256 * 4);
  (void)ws_size; (void)n_in; (void)out_size;

  const dim3 blk(256);

  // ---- degrees / symmetric normalization (self-loop counts as deg 1)
  fill_f32<<<1024, blk, 0, stream>>>(dinv, 1.0f, (long)N);
  edge_deg<<<CDIV(E, 256), blk, 0, stream>>>(edst, dinv, E);
  deg_to_dinv<<<CDIV(N, 256), blk, 0, stream>>>(dinv, N);
  edge_norm<<<CDIV(E, 256), blk, 0, stream>>>(esrc, edst, dinv, enorm, E);

  // ---- layer 1: transform (trivial K=1), self-loop init, edge scatter
  l1_transform<<<CDIV(N * 64, 256), blk, 0, stream>>>(x, W1, P, N);
  self_init<<<CDIV(N * 64, 256), blk, 0, stream>>>(P, dinv, Q, N * 64, 6);
  edge_scatter<<<(unsigned)CDIV((long)E * 16, 256), blk, 0, stream>>>(
      P, Q, esrc, edst, enorm, (long)E * 16, 4, 15, 64);

  // ---- layer 2: GEMM (fused relu(Q+b1)) -> P[N,128]; scatter P->Q
  wmma_gemm_bf16<<<dim3(N / 64, 128 / 64), blk, 0, stream>>>(Q, W2, P, N, 128, 64, b1);
  self_init<<<CDIV(N * 128, 256), blk, 0, stream>>>(P, dinv, Q, N * 128, 7);
  edge_scatter<<<(unsigned)CDIV((long)E * 32, 256), blk, 0, stream>>>(
      P, Q, esrc, edst, enorm, (long)E * 32, 5, 31, 128);

  // ---- layer 3: GEMM (fused relu(Q+b2)) -> P[N,256]; scatter P->Q (= agg3)
  wmma_gemm_bf16<<<dim3(N / 64, 256 / 64), blk, 0, stream>>>(Q, W3, P, N, 256, 128, b2);
  self_init<<<CDIV(N * 256, 256), blk, 0, stream>>>(P, dinv, Q, N * 256, 8);
  edge_scatter<<<(unsigned)CDIV((long)E * 64, 256), blk, 0, stream>>>(
      P, Q, esrc, edst, enorm, (long)E * 64, 6, 63, 256);
  // Q now holds agg3 (pre-bias/relu); h3 = relu(Q + b3) is fused downstream.

  // ---- gate MLP: G1raw = relu(Q+b3)@g1 -> P ; G2raw = relu(P+gb1)@g2 -> R
  wmma_gemm_bf16<<<dim3(N / 64, 256 / 64), blk, 0, stream>>>(Q, g1, P, N, 256, 256, b3);
  wmma_gemm_bf16<<<dim3(N / 64, 256 / 64), blk, 0, stream>>>(P, g2, R, N, 256, 256, gb1);
  rowdot_relu<<<CDIV(N, 8), blk, 0, stream>>>(R, gb2, g3, gb3, ge, N, 256);

  // ---- segment softmax over sorted batch
  fill_f32<<<64, blk, 0, stream>>>(mbuf, -1e30f, (long)NGRAPHS);
  fill_f32<<<64, blk, 0, stream>>>(zbuf, 0.0f, (long)NGRAPHS);
  seg_max<<<CDIV(N, 256), blk, 0, stream>>>(ge, batch, mbuf, N);
  exp_z<<<CDIV(N, 256), blk, 0, stream>>>(ge, batch, mbuf, zbuf, N);

  // ---- attention-weighted pooling into [NGRAPHS, 256]
  fill_f32<<<1024, blk, 0, stream>>>(pooled, 0.0f, (long)NGRAPHS * 256);
  pooled_scatter<<<CDIV(N * 64, 256), blk, 0, stream>>>(Q, b3, ge, zbuf, batch, pooled, N);

  // ---- head MLP: pooled@h1 -> P[8192,128]; relu(P+hb1)@h2 -> R[8192,64]; final dot
  wmma_gemm_bf16<<<dim3(NGRAPHS / 64, 128 / 64), blk, 0, stream>>>(
      pooled, h1, P, NGRAPHS, 128, 256, nullptr);
  wmma_gemm_bf16<<<dim3(NGRAPHS / 64, 64 / 64), blk, 0, stream>>>(
      P, h2, R, NGRAPHS, 64, 128, hb1);
  rowdot_relu<<<CDIV(NGRAPHS, 8), blk, 0, stream>>>(R, hb2, h3, hb3, out, NGRAPHS, 64);
}
